// En_DecoderGAT_55594056679683
// MI455X (gfx1250) — matrix operand order repaired and checked by reading, not compiled, not run on hardware
//
#include <hip/hip_runtime.h>

typedef __attribute__((ext_vector_type(16))) _Float16 v16h;
typedef __attribute__((ext_vector_type(8)))  _Float16 v8h;
typedef __attribute__((ext_vector_type(4)))  _Float16 v4h;
typedef __attribute__((ext_vector_type(8)))  float    v8f;
typedef __attribute__((ext_vector_type(4)))  float    v4f;

union V16H { v16h v; v8h h[2]; };

// ---- CDNA5 async global->LDS copy (ASYNCcnt-tracked), per CDNA5 ISA ----
__device__ __forceinline__ void async_g2l_b128(const float* g, float* l) {
  unsigned lds = (unsigned)(unsigned long long)(void*)l;   // generic LDS addr truncates to LDS offset
  asm volatile("global_load_async_to_lds_b128 %0, %1, off" :: "v"(lds), "v"(g) : "memory");
}
#define WAIT_ASYNC_1() asm volatile("s_wait_asynccnt 1" ::: "memory")
#define WAIT_ASYNC_0() asm volatile("s_wait_asynccnt 0" ::: "memory")

// ---------------- elementwise casts ----------------
__global__ void cast_f16_kernel(const float* __restrict__ x, _Float16* __restrict__ y, int n) {
  int i = blockIdx.x * blockDim.x + threadIdx.x;
  int stride = gridDim.x * blockDim.x;
  for (; i < n; i += stride) y[i] = (_Float16)x[i];
}

// W[K,N] row-major  ->  WT[N,K] f16 (so WMMA B-fragments are K-contiguous per lane)
__global__ void cast_transpose_f16_kernel(const float* __restrict__ W, _Float16* __restrict__ WT,
                                          int K, int N) {
  int i = blockIdx.x * blockDim.x + threadIdx.x;
  if (i >= K * N) return;
  int k = i / N, n = i - k * N;
  WT[(size_t)n * K + k] = (_Float16)W[i];
}

// ---------------- per-row attention score dots: s = Wh@a[:F], t = Wh@a[F:] ----------------
__global__ void scores_kernel(const float* __restrict__ Wh, const float* __restrict__ a,
                              float* __restrict__ s, float* __restrict__ t, int Nn, int F) {
  int row  = blockIdx.x * (blockDim.x >> 5) + (threadIdx.x >> 5);
  int lane = threadIdx.x & 31;
  if (row >= Nn) return;
  const float* wr = Wh + (size_t)row * F;
  float ps = 0.0f, pt = 0.0f;
  for (int c = lane; c < F; c += 32) {
    float w = wr[c];
    ps += w * a[c];
    pt += w * a[F + c];
  }
  #pragma unroll
  for (int off = 16; off > 0; off >>= 1) {
    ps += __shfl_down(ps, off, 32);
    pt += __shfl_down(pt, off, 32);
  }
  if (lane == 0) { s[row] = ps; t[row] = pt; }
}

// ---------------- WMMA GEMM: C[M,N]f32 = A[M,K]f16 @ BT[N,K]f16^T; also writes CT[N,M]f16 ----------------
// block = 128 threads (4 waves); block tile 64x64; wave tile 16 rows x 64 cols (4 accumulators).
__global__ void __launch_bounds__(128)
gemm_wmma_f16_kernel(const _Float16* __restrict__ A,   // [M,K] row-major
                     const _Float16* __restrict__ BT,  // [N,K] row-major (B transposed)
                     float* __restrict__ C,            // [M,N] f32
                     _Float16* __restrict__ CT,        // [N,M] f16 (transposed copy for next stage)
                     int M, int N, int K) {
  const int lane = threadIdx.x & 31;
  const int wave = threadIdx.x >> 5;
  const int lp = lane & 15, hi = lane >> 4;
  const int m0 = blockIdx.y * 64 + wave * 16;
  const int n0 = blockIdx.x * 64;

  v8f acc[4] = {};
  // A fragment (16x32 f16): lane<16 -> row=lp, K = k0+{0..7, 16..23}; lane>=16 -> K = k0+{8..15, 24..31}
  const _Float16* aRow = A + (size_t)(m0 + lp) * K + hi * 8;

  for (int k0 = 0; k0 < K; k0 += 32) {
    V16H af;
    af.h[0] = *(const v8h*)(aRow + k0);
    af.h[1] = *(const v8h*)(aRow + k0 + 16);
    if (k0 + 32 < K) __builtin_prefetch(aRow + k0 + 32, 0, 1);   // global_prefetch_b8
    #pragma unroll
    for (int t4 = 0; t4 < 4; ++t4) {
      // B fragment (32x16): lane<16 -> col=n, K=k0+{0..7, 16..23} ; lane>=16 -> K=k0+{8..15, 24..31}
      const _Float16* bp = BT + (size_t)(n0 + t4 * 16 + lp) * K + k0 + hi * 8;
      V16H bf;
      bf.h[0] = *(const v8h*)(bp);
      bf.h[1] = *(const v8h*)(bp + 16);
      acc[t4] = __builtin_amdgcn_wmma_f32_16x16x32_f16(false, af.v, false, bf.v,
                                                       (short)0, acc[t4], false, false);
    }
  }
  #pragma unroll
  for (int t4 = 0; t4 < 4; ++t4) {
    const int n = n0 + t4 * 16 + lp;
    #pragma unroll
    for (int r = 0; r < 8; ++r) {                  // C layout: VGPR r -> row m0 + r + 8*hi
      const int m = m0 + r + 8 * hi;
      float v = acc[t4][r];
      C [(size_t)m * N + n] = v;
      CT[(size_t)n * M + m] = (_Float16)v;
    }
  }
}

// ---------------- masked-softmax aggregation: out = softmax_row(mask(sigmoid(s_i+t_j))) @ Wh ----------------
// adj streamed HBM->LDS with double-buffered async loads. The 16x32 attention-weight tile is built
// COOPERATIVELY by all 128 threads (4 weights / thread -> 4x fewer transcendentals than per-wave
// redundant compute) and stored in LDS in the exact WMMA A-fragment per-lane layout, so each wave
// fetches its A fragment with two ds_load_b128. Wh is consumed as B-fragments from the pre-transposed
// f16 WhT (L2-resident). block = 128 threads; 16 rows per block; wave w covers cols [w*64, w*64+64).
__global__ void __launch_bounds__(128)
gat_aggregate_kernel(const float* __restrict__ adj,       // [Nn,Nn] f32
                     const _Float16* __restrict__ WhT,    // [F,Nn] f16
                     const float* __restrict__ svec,      // [Nn]
                     const float* __restrict__ tvec,      // [Nn]
                     float* __restrict__ outMain,         // [Nn,F]
                     float* __restrict__ outElu,          // [Nn,F] or null
                     _Float16* __restrict__ outH16,       // [Nn,F] f16 of elu, or null
                     int Nn, int F, int eluMain) {
  __shared__ __align__(16) float    sAdj[2][16 * 32];
  __shared__ __align__(16) _Float16 sWt [2][32 * 16];     // [lane*16 + h]: A-fragment layout
  __shared__ float sPart[128];
  const int lane = threadIdx.x & 31;
  const int wave = threadIdx.x >> 5;
  const int lp = lane & 15, hi = lane >> 4;
  const int m0 = blockIdx.x * 16;
  const int n0 = wave * 64;
  const int S  = Nn >> 5;                          // 256 k-steps of 32 neighbors

  // this thread's 4-entry slice of the A fragment: h in [hb, hb+4)
  const int hb = wave * 4;
  // fragment h -> adj column: h<8 -> hi*8+h ; h>=8 -> 16+hi*8+(h-8)
  const int colbase = (hb < 8) ? (hi * 8 + hb) : (16 + hi * 8 + (hb - 8));

  // cooperative async staging map: thread -> 4 consecutive floats of the 16x32 adj tile
  const int sr = threadIdx.x >> 3;                 // 0..15
  const int sc = (threadIdx.x & 7) << 2;           // 0..28
  const float* gstage = adj + (size_t)(m0 + sr) * Nn + sc;

  // prologue: prefetch adj tiles 0 and 1
  async_g2l_b128(gstage +  0, &sAdj[0][sr * 32 + sc]);
  async_g2l_b128(gstage + 32, &sAdj[1][sr * 32 + sc]);

  const float srow = svec[m0 + lp];                // score s_i for this thread's row
  v8f acc[4] = {};
  float psum = 0.0f;                               // partial row sum of weights

  for (int step = 0; step < S; ++step) {
    const int jb = step << 5;
    if (step + 2 <= S) { WAIT_ASYNC_1(); } else { WAIT_ASYNC_0(); }
    __syncthreads();                               // adj tile `step` visible to all waves

    // phase A: block cooperatively builds weight tile (2 transcendentals per weight, 4 weights/thread)
    v4f av = *(const v4f*)&sAdj[step & 1][lp * 32 + colbase];
    v4f tv = *(const v4f*)(tvec + jb + colbase);
    v4h wv;
    #pragma unroll
    for (int i = 0; i < 4; ++i) {
      float xs  = srow + tv[i];
      float sig = 1.0f / (1.0f + __expf(-xs));
      float w   = (av[i] > 0.0f) ? __expf(sig) : 0.0f;   // masked exp(sigmoid); shift-free softmax
      wv[i] = (_Float16)w;
      psum += w;
    }
    *(v4h*)&sWt[step & 1][lane * 16 + hb] = wv;
    __syncthreads();                               // weight tile ready; sAdj[step&1] consumed

    // earliest safe point to overwrite sAdj[step&1] with tile step+2
    if (step + 2 < S)
      async_g2l_b128(gstage + (size_t)jb + 64, &sAdj[step & 1][sr * 32 + sc]);

    // phase B: A fragment from LDS, B fragments from L2-resident WhT, 4 WMMAs
    V16H af;
    af.h[0] = *(const v8h*)&sWt[step & 1][lane * 16];
    af.h[1] = *(const v8h*)&sWt[step & 1][lane * 16 + 8];
    #pragma unroll
    for (int t4 = 0; t4 < 4; ++t4) {
      const _Float16* bp = WhT + (size_t)(n0 + t4 * 16 + lp) * Nn + jb + hi * 8;
      V16H bf;
      bf.h[0] = *(const v8h*)(bp);
      bf.h[1] = *(const v8h*)(bp + 16);
      acc[t4] = __builtin_amdgcn_wmma_f32_16x16x32_f16(false, af.v, false, bf.v,
                                                       (short)0, acc[t4], false, false);
    }
  }

  // softmax denominators: rowsum[m] = sum over the 8 threads (tid = m + 16k) that touched row m
  sPart[threadIdx.x] = psum;
  __syncthreads();
  float rs[8];
  #pragma unroll
  for (int r = 0; r < 8; ++r) {
    const int m = r + 8 * hi;
    float a = 0.0f;
    #pragma unroll
    for (int k = 0; k < 8; ++k) a += sPart[m + 16 * k];
    rs[r] = a;
  }

  #pragma unroll
  for (int t4 = 0; t4 < 4; ++t4) {
    const int n = n0 + t4 * 16 + lp;
    #pragma unroll
    for (int r = 0; r < 8; ++r) {                  // C layout: VGPR r -> row m0 + r + 8*hi
      const int m = m0 + r + 8 * hi;
      float v = acc[t4][r] / rs[r];
      float e = (v > 0.0f) ? v : (__expf(v) - 1.0f);   // ELU
      outMain[(size_t)m * F + n] = eluMain ? e : v;
      if (outElu)  outElu [(size_t)m * F + n] = e;
      if (outH16)  outH16 [(size_t)m * F + n] = (_Float16)e;
    }
  }
}

// ---------------- host-side launcher ----------------
extern "C" void kernel_launch(void* const* d_in, const int* in_sizes, int n_in,
                              void* d_out, int out_size, void* d_ws, size_t ws_size,
                              hipStream_t stream) {
  const int Nn = 8192, IN = 512, HID = 256, OUT = 256;
  const float* adj = (const float*)d_in[0];
  const float* x   = (const float*)d_in[1];
  const float* W0  = (const float*)d_in[2];
  const float* a0  = (const float*)d_in[3];
  const float* W1  = (const float*)d_in[4];
  const float* a1  = (const float*)d_in[5];

  float* out   = (float*)d_out;
  float* h2    = out;                              // [Nn,OUT]
  float* h1    = out + (size_t)Nn * OUT;           // [Nn,HID]
  float* h2elu = out + 2 * (size_t)Nn * OUT;       // [Nn,OUT]

  char* p = (char*)d_ws;
  auto alloc = [&](size_t bytes) -> char* { char* r = p; p += (bytes + 255) & ~(size_t)255; return r; };
  _Float16* xh   = (_Float16*)alloc((size_t)Nn * IN  * 2);
  _Float16* W0T  = (_Float16*)alloc((size_t)HID * IN * 2);
  _Float16* W1T  = (_Float16*)alloc((size_t)OUT * HID * 2);
  float*    Wh0  = (float*)   alloc((size_t)Nn * HID * 4);
  _Float16* Wh0T = (_Float16*)alloc((size_t)HID * Nn * 2);
  _Float16* h1h  = (_Float16*)alloc((size_t)Nn * HID * 2);
  float*    Wh1  = (float*)   alloc((size_t)Nn * OUT * 4);
  _Float16* Wh1T = (_Float16*)alloc((size_t)OUT * Nn * 2);
  float* s0 = (float*)alloc((size_t)Nn * 4);
  float* t0 = (float*)alloc((size_t)Nn * 4);
  float* s1 = (float*)alloc((size_t)Nn * 4);
  float* t1 = (float*)alloc((size_t)Nn * 4);

  // precision conversion
  cast_f16_kernel<<<4096, 256, 0, stream>>>(x, xh, Nn * IN);
  cast_transpose_f16_kernel<<<(IN * HID + 255) / 256, 256, 0, stream>>>(W0, W0T, IN, HID);
  cast_transpose_f16_kernel<<<(HID * OUT + 255) / 256, 256, 0, stream>>>(W1, W1T, HID, OUT);

  // layer 0: Wh0 = x@W0 ; s0,t0 ; h1 = elu(att@Wh0)
  gemm_wmma_f16_kernel<<<dim3(HID / 64, Nn / 64), 128, 0, stream>>>(xh, W0T, Wh0, Wh0T, Nn, HID, IN);
  scores_kernel<<<Nn / 8, 256, 0, stream>>>(Wh0, a0, s0, t0, Nn, HID);
  gat_aggregate_kernel<<<Nn / 16, 128, 0, stream>>>(adj, Wh0T, s0, t0, h1, nullptr, h1h, Nn, HID, 1);

  // layer 1: Wh1 = h1@W1 ; s1,t1 ; h2 = att@Wh1, plus elu(h2)
  gemm_wmma_f16_kernel<<<dim3(OUT / 64, Nn / 64), 128, 0, stream>>>(h1h, W1T, Wh1, Wh1T, Nn, OUT, HID);
  scores_kernel<<<Nn / 8, 256, 0, stream>>>(Wh1, a1, s1, t1, Nn, OUT);
  gat_aggregate_kernel<<<Nn / 16, 128, 0, stream>>>(adj, Wh1T, s1, t1, h2, h2elu, nullptr, Nn, OUT, 0);

  (void)in_sizes; (void)n_in; (void)out_size; (void)ws_size;
}
